// S5Layer_90512140796632
// MI455X (gfx1250) — matrix operand order repaired and checked by reading, not compiled, not run on hardware
//
#include <hip/hip_runtime.h>
#include <hip/hip_bf16.h>

typedef __attribute__((ext_vector_type(16))) _Float16 v16h;
typedef __attribute__((ext_vector_type(8)))  _Float16 v8h;
typedef __attribute__((ext_vector_type(8)))  float    v8f;

#define M_TOTAL 32768   // BATCH*LENGTH tokens
#define K_TOTAL 512     // inner dim (H for GEMM1, 2P for GEMM2)
#define N_TOTAL 512     // output dim (2P for GEMM1, H for GEMM2)
#define P_DIM   256
#define SEQ_LEN 4096

#define NBLK        128          // N columns per block
#define LDS_STRIDE  520          // halves per LDS row: 260 dwords, 260%64=4
                                 // -> half-wave ds_load_b128 is bank-conflict-free

// ---------------------------------------------------------------------------
// Prep: build f16 weight matrices, N-major / K-contiguous (WMMA B-operand order)
//   W1T[n][k]: n<256 -> Bbar_re[p=n][h=k] ; n>=256 -> Bbar_im[p=n-256][h=k]
//   W2T[h][k]: k<256 -> 2*C_re[h][p=k]    ; k>=256 -> -2*C_im[h][p=k-256]
// ---------------------------------------------------------------------------
__global__ __launch_bounds__(256) void prep_kernel(
    const float* __restrict__ Lre, const float* __restrict__ Lim,
    const float* __restrict__ log_dt,
    const float* __restrict__ B_comps, const float* __restrict__ C_comps,
    _Float16* __restrict__ W1T, _Float16* __restrict__ W2T)
{
    int idx = blockIdx.x * 256 + threadIdx.x;      // 0 .. 2*512*512-1
    int which = idx >> 18;
    int r = (idx & 262143) >> 9;                   // row n (or h), 0..511
    int k = idx & 511;                             // col k, 0..511
    if (which == 0) {
        int p = r & 255;
        float dt = __expf(log_dt[p]);
        float lr = Lre[p], li = Lim[p];
        float er = lr * dt, ei = li * dt;
        float mag = __expf(er);
        float abr = mag * __cosf(ei);              // Lambda_bar
        float abi = mag * __sinf(ei);
        float den = lr * lr + li * li;
        float gr = ((abr - 1.f) * lr + abi * li) / den;   // g=(Lb-1)/Lambda
        float gi = (abi * lr - (abr - 1.f) * li) / den;
        float b0 = B_comps[p * 512 + k];
        float b1 = B_comps[P_DIM * 512 + p * 512 + k];
        float v = (r < P_DIM) ? (gr * b0 - gi * b1) : (gr * b1 + gi * b0);
        W1T[r * 512 + k] = (_Float16)v;
    } else {
        int h = r;
        float v;
        if (k < P_DIM) v =  2.f * C_comps[h * P_DIM + k];
        else           v = -2.f * C_comps[512 * P_DIM + h * P_DIM + (k - P_DIM)];
        W2T[h * 512 + k] = (_Float16)v;
    }
}

// ---------------------------------------------------------------------------
// Fragment loaders: A and B both use contiguous k order (k = k0 + 16*hi + e).
// ---------------------------------------------------------------------------
template <typename AT>
__device__ __forceinline__ v16h load_a_frag(const AT* __restrict__ arow,
                                            int k0, int hi)
{
    // A 16x32: lanes<16 hold K = [k0..k0+15];
    //          lanes>=16 hold K = [k0+16..k0+31]. One contiguous run per lane.
    v16h af;
#pragma unroll
    for (int e = 0; e < 8; ++e)
        af[e] = (_Float16)arow[k0 + hi * 16 + e];
#pragma unroll
    for (int e = 0; e < 8; ++e)
        af[8 + e] = (_Float16)arow[k0 + hi * 16 + 8 + e];
    return af;
}

__device__ __forceinline__ v16h load_b_lds(const _Float16* __restrict__ sW,
                                           int nLocal, int k0, int hi)
{
    // B 32x16: lane = column n, elements = contiguous K = k0 + hi*16 + e.
    // 16B-aligned LDS rows -> two ds_load_b128 per fragment.
    const _Float16* p = sW + nLocal * LDS_STRIDE + k0 + hi * 16;
    v8h lo = *reinterpret_cast<const v8h*>(p);
    v8h hp = *reinterpret_cast<const v8h*>(p + 8);
    v16h bf;
#pragma unroll
    for (int e = 0; e < 8; ++e) { bf[e] = lo[e]; bf[8 + e] = hp[e]; }
    return bf;
}

// ---------------------------------------------------------------------------
// WMMA GEMM: Out[M x N] = A[M x K] * W^T  (W stored [N][K], K-contiguous)
// Block = 256 threads = 8 waves; wave tile 16(M) x 64(N); block 64(M) x 128(N).
// Weight slab (128 x 512 f16 = 128KB) staged once into LDS, K-loop feeds B
// operands from LDS (short ds waits) while A global loads are prefetched one
// K-step ahead in registers.
// MODE 0: store fp32 transposed Out[n][m]   (Bu for the scan, stream-friendly)
// MODE 1: store fp32 Out[m][h] = acc + D[h]*U[m][h]
// ---------------------------------------------------------------------------
template <typename AT, int MODE>
__global__ __launch_bounds__(256) void gemm_wmma_kernel(
    const AT* __restrict__ A, const _Float16* __restrict__ W,
    float* __restrict__ Out, const float* __restrict__ Dvec,
    const float* __restrict__ U)
{
    extern __shared__ _Float16 sW[];               // NBLK * LDS_STRIDE halves

    const int tid    = threadIdx.x;
    const int lane   = tid & 31;
    const int waveId = tid >> 5;
    const int waveM  = waveId & 3;
    const int waveN  = waveId >> 2;
    const int hi     = lane >> 4;       // half-wave select
    const int ln     = lane & 15;
    const int mW = blockIdx.x * 64 + waveM * 16;   // wave M base
    const int nB = blockIdx.y * NBLK;              // block N base

    // ---- Stage weight slab: NBLK rows x 512 halves, 16B chunks -------------
    {
        const _Float16* src = W + (size_t)nB * K_TOTAL;
        for (int c = tid; c < NBLK * 64; c += 256) {   // 64 chunks of 8 halves/row
            int row = c >> 6;
            int kp  = (c & 63) * 8;
            v8h v = *reinterpret_cast<const v8h*>(src + row * K_TOTAL + kp);
            *reinterpret_cast<v8h*>(&sW[row * LDS_STRIDE + kp]) = v;
        }
    }
    __syncthreads();

    v8f acc[4] = {};
    const AT* arow = A + (size_t)(mW + ln) * K_TOTAL;
    const int nL = waveN * 64 + ln;                // wave's local B row base

    // Prologue: A fragment for k0 = 0
    v16h af = load_a_frag(arow, 0, hi);

#pragma unroll
    for (int k0 = 0; k0 < K_TOTAL; k0 += 32) {
        v16h af_n;
        const int kn = k0 + 32;
        if (kn < K_TOTAL)
            af_n = load_a_frag(arow, kn, hi);      // prefetch next A (global)

        v16h bf0 = load_b_lds(sW, nL +  0, k0, hi);
        v16h bf1 = load_b_lds(sW, nL + 16, k0, hi);
        v16h bf2 = load_b_lds(sW, nL + 32, k0, hi);
        v16h bf3 = load_b_lds(sW, nL + 48, k0, hi);

        acc[0] = __builtin_amdgcn_wmma_f32_16x16x32_f16(
            false, af, false, bf0, (short)0, acc[0], false, false);
        acc[1] = __builtin_amdgcn_wmma_f32_16x16x32_f16(
            false, af, false, bf1, (short)0, acc[1], false, false);
        acc[2] = __builtin_amdgcn_wmma_f32_16x16x32_f16(
            false, af, false, bf2, (short)0, acc[2], false, false);
        acc[3] = __builtin_amdgcn_wmma_f32_16x16x32_f16(
            false, af, false, bf3, (short)0, acc[3], false, false);

        if (kn < K_TOTAL) af = af_n;
    }

    // C/D layout: lane ln = column N, VGPR i = row M = hi*8 + i
    if (MODE == 0) {
#pragma unroll
        for (int nt = 0; nt < 4; ++nt) {
            int n = nB + waveN * 64 + nt * 16 + ln;
            float* dst = Out + (size_t)n * M_TOTAL + mW + hi * 8;
#pragma unroll
            for (int i = 0; i < 8; ++i) dst[i] = acc[nt][i];  // contiguous 32B
        }
    } else {
#pragma unroll
        for (int nt = 0; nt < 4; ++nt) {
            int h = nB + waveN * 64 + nt * 16 + ln;
            float dh = Dvec[h];
#pragma unroll
            for (int i = 0; i < 8; ++i) {
                int m = mW + hi * 8 + i;
                size_t off = (size_t)m * N_TOTAL + h;
                Out[off] = acc[nt][i] + dh * U[off];
            }
        }
    }
}

// ---------------------------------------------------------------------------
// Chunked parallel scan: x_l = a*x_{l-1} + Bu_l, a = Lambda_bar[p] (complex)
// Grid: 256 blocks (8 batches x 32 p-groups); block = 8 waves (one p each),
// lane = chunk of 128 timesteps. 3 phases: local scan, LDS Hillis-Steele on
// (a^128, f) carry pairs, replay with carry writing f16 X[m][{p, 256+p}].
// ---------------------------------------------------------------------------
__global__ __launch_bounds__(256) void scan_kernel(
    const float* __restrict__ BuT,
    const float* __restrict__ Lre, const float* __restrict__ Lim,
    const float* __restrict__ log_dt, _Float16* __restrict__ X)
{
    const int lane = threadIdx.x & 31;
    const int wave = threadIdx.x >> 5;
    const int b = blockIdx.x >> 5;
    const int p = (blockIdx.x & 31) * 8 + wave;

    float dt = __expf(log_dt[p]);
    float er = Lre[p] * dt, ei = Lim[p] * dt;
    float mag  = __expf(er);
    float a_re = mag * __cosf(ei);
    float a_im = mag * __sinf(ei);

    const int CH = 128;
    const int m0 = b * SEQ_LEN + lane * CH;
    const float* bur = BuT + (size_t)p * M_TOTAL + m0;
    const float* bui = BuT + (size_t)(P_DIM + p) * M_TOTAL + m0;

    // Phase 1: local scan from zero state
    float xr = 0.f, xi = 0.f;
    for (int i = 0; i < CH; ++i) {
        float br = bur[i], bi = bui[i];
        float nr = xr * a_re - xi * a_im + br;
        float ni = xr * a_im + xi * a_re + bi;
        xr = nr; xi = ni;
    }

    // a^128 via 7 complex squarings
    float Ar = a_re, Ai = a_im;
#pragma unroll
    for (int s = 0; s < 7; ++s) {
        float nr = Ar * Ar - Ai * Ai;
        float ni = 2.f * Ar * Ai;
        Ar = nr; Ai = ni;
    }

    // Phase 2: Hillis-Steele over the 32 chunk carries (per p-row)
    __shared__ float sAr[8][32], sAi[8][32], sFr[8][32], sFi[8][32];
    sAr[wave][lane] = Ar; sAi[wave][lane] = Ai;
    sFr[wave][lane] = xr; sFi[wave][lane] = xi;
    __syncthreads();
    for (int s = 1; s < 32; s <<= 1) {
        float pAr = 0.f, pAi = 0.f, pFr = 0.f, pFi = 0.f;
        bool act = (lane >= s);
        if (act) {
            pAr = sAr[wave][lane - s]; pAi = sAi[wave][lane - s];
            pFr = sFr[wave][lane - s]; pFi = sFi[wave][lane - s];
        }
        __syncthreads();
        if (act) {
            float cAr = sAr[wave][lane], cAi = sAi[wave][lane];
            float cFr = sFr[wave][lane], cFi = sFi[wave][lane];
            sAr[wave][lane] = cAr * pAr - cAi * pAi;
            sAi[wave][lane] = cAr * pAi + cAi * pAr;
            sFr[wave][lane] = cAr * pFr - cAi * pFi + cFr;
            sFi[wave][lane] = cAr * pFi + cAi * pFr + cFi;
        }
        __syncthreads();
    }
    float cr = 0.f, ci = 0.f;
    if (lane > 0) { cr = sFr[wave][lane - 1]; ci = sFi[wave][lane - 1]; }

    // Phase 3: replay with incoming carry, emit f16 A-matrix for GEMM2
    xr = cr; xi = ci;
    for (int i = 0; i < CH; ++i) {
        float br = bur[i], bi = bui[i];
        float nr = xr * a_re - xi * a_im + br;
        float ni = xr * a_im + xi * a_re + bi;
        xr = nr; xi = ni;
        size_t m = (size_t)(m0 + i);
        X[m * K_TOTAL + p]         = (_Float16)xr;
        X[m * K_TOTAL + P_DIM + p] = (_Float16)xi;
    }
}

// ---------------------------------------------------------------------------
// Host launcher. inputs order: inputs, Lambda_re, Lambda_im, B_comps, C_comps,
// D, log_dt. Workspace: W1T f16 | W2T f16 | BuT f32 [512][32768] | X f16.
// ---------------------------------------------------------------------------
extern "C" void kernel_launch(void* const* d_in, const int* in_sizes, int n_in,
                              void* d_out, int out_size, void* d_ws, size_t ws_size,
                              hipStream_t stream) {
    const float* inputs  = (const float*)d_in[0];
    const float* Lre     = (const float*)d_in[1];
    const float* Lim     = (const float*)d_in[2];
    const float* B_comps = (const float*)d_in[3];
    const float* C_comps = (const float*)d_in[4];
    const float* Dvec    = (const float*)d_in[5];
    const float* log_dt  = (const float*)d_in[6];
    float* out = (float*)d_out;

    char* ws = (char*)d_ws;
    const size_t wbytes = (size_t)512 * 512 * sizeof(_Float16);        // 512 KB
    _Float16* W1T = (_Float16*)ws;
    _Float16* W2T = (_Float16*)(ws + wbytes);
    float*    BuT = (float*)(ws + 2 * wbytes);
    _Float16* X   = (_Float16*)(ws + 2 * wbytes +
                                (size_t)N_TOTAL * M_TOTAL * sizeof(float));

    // 1) weights (2*512*512 elements / 256 threads)
    prep_kernel<<<2048, 256, 0, stream>>>(Lre, Lim, log_dt, B_comps, C_comps,
                                          W1T, W2T);

    const size_t lds_bytes = (size_t)NBLK * LDS_STRIDE * sizeof(_Float16); // 130 KB

    // 2) Bu = u @ W1T^T, stored transposed [n][m]
    dim3 gemm_grid(M_TOTAL / 64, N_TOTAL / NBLK);
    gemm_wmma_kernel<float, 0><<<gemm_grid, 256, lds_bytes, stream>>>(
        inputs, W1T, BuT, nullptr, nullptr);

    // 3) recurrence scan -> X f16 [m][2P]
    scan_kernel<<<256, 256, 0, stream>>>(BuT, Lre, Lim, log_dt, X);

    // 4) out = X @ W2T^T + D*u
    gemm_wmma_kernel<_Float16, 1><<<gemm_grid, 256, lds_bytes, stream>>>(
        X, W2T, out, Dvec, inputs);
}
